// QuantumPerceptron_13365938225673
// MI455X (gfx1250) — matrix operand order repaired and checked by reading, not compiled, not run on hardware
//
#include <hip/hip_runtime.h>
#include <hip/hip_bf16.h>
#include <math.h>

typedef _Float16 v16h __attribute__((ext_vector_type(16)));
typedef float    v8f  __attribute__((ext_vector_type(8)));

#define DIMQ 128
#define NMAT (DIMQ * DIMQ)   // 16384 complex elements per 128x128
#define TQ 5
#define LQ 4
#define NQ 7
#define SCALE_POW 8          // expm scaling 2^-8
#define NTERMS 12            // Taylor terms

__device__ __forceinline__ float2 cmulf2(float2 a, float2 b) {
    return make_float2(fmaf(a.x, b.x, -a.y * b.y), fmaf(a.x, b.y, a.y * b.x));
}

// ---------------------------------------------------------------------------
// 1) As[t] = (-i * t_time / 2^S) * (m1 - m2 + m3)   (complex f32, re part = 0)
// ---------------------------------------------------------------------------
__global__ void build_as_kernel(float2* __restrict__ As,
                                const float* __restrict__ m1,
                                const float* __restrict__ m2,
                                const float* __restrict__ m3) {
    int gid = blockIdx.x * 256 + threadIdx.x;
    if (gid >= NMAT) return;
    float m = m1[gid] - m2[gid] + m3[gid];
    const float times[TQ] = {0.1f, 0.3f, 0.5f, 0.7f, 0.9f};
    const float inv = 1.0f / 256.0f;   // 2^-SCALE_POW
#pragma unroll
    for (int t = 0; t < TQ; ++t)
        As[(size_t)t * NMAT + gid] = make_float2(0.f, -times[t] * inv * m);
}

// ---------------------------------------------------------------------------
// 2) Per-qubit gates  G = Rz(ga) @ Rx(be) @ Rz(a)  -> (L,2,N) 2x2 complex
// ---------------------------------------------------------------------------
__global__ void build_gates_kernel(float2* __restrict__ G,
                                   const float* __restrict__ params) {
    int tid = threadIdx.x;
    if (tid >= LQ * 2 * NQ) return;
    int q = tid % NQ;
    int w = (tid / NQ) % 2;
    int l = tid / (2 * NQ);
    float a  = params[((0 * LQ + l) * 2 + w) * NQ + q];
    float be = params[((1 * LQ + l) * 2 + w) * NQ + q];
    float ga = params[((2 * LQ + l) * 2 + w) * NQ + q];
    float ca = cosf(a),  sa = sinf(a);
    float cb = cosf(be), sb = sinf(be);
    float cg = cosf(ga), sg = sinf(ga);
    float2 d0 = make_float2(ca, -sa), d1 = make_float2(ca, sa);
    float2 e0 = make_float2(cg, -sg), e1 = make_float2(cg, sg);
    float2 misb = make_float2(0.f, -sb);
    float2 g00 = cmulf2(e0, make_float2(cb * d0.x, cb * d0.y));
    float2 g01 = cmulf2(e0, cmulf2(misb, d1));
    float2 g10 = cmulf2(e1, cmulf2(misb, d0));
    float2 g11 = cmulf2(e1, make_float2(cb * d1.x, cb * d1.y));
    float2* o = G + (size_t)(l * 2 + w) * NQ * 4 + q * 4;
    o[0] = g00; o[1] = g01; o[2] = g10; o[3] = g11;
}

// ---------------------------------------------------------------------------
// 3) Full kron-chain unitaries: U[idx][i][j] = prod_q G[idx][q][bit_q(i)][bit_q(j)]
//    idx = l*2 + w  (w=0 -> U1[l], w=1 -> U2[l]); qubit 0 most significant.
// ---------------------------------------------------------------------------
__global__ void build_u_kernel(float2* __restrict__ U, const float2* __restrict__ G) {
    int z = blockIdx.z;
    int j = blockIdx.x * 16 + threadIdx.x;
    int i = blockIdx.y * 16 + threadIdx.y;
    const float2* g = G + (size_t)z * NQ * 4;
    float2 acc = g[(((i >> 6) & 1) * 2 + ((j >> 6) & 1))];
#pragma unroll
    for (int q = 1; q < NQ; ++q) {
        int bi = (i >> (6 - q)) & 1;
        int bj = (j >> (6 - q)) & 1;
        acc = cmulf2(acc, g[q * 4 + bi * 2 + bj]);
    }
    U[(size_t)z * NMAT + i * DIMQ + j] = acc;
}

// ---------------------------------------------------------------------------
// 4) ACC = TERM = I  (per t)
// ---------------------------------------------------------------------------
__global__ void zinit_eye_kernel(float2* __restrict__ Acc, float2* __restrict__ Term) {
    int gid = blockIdx.x * 256 + threadIdx.x;
    if (gid >= TQ * NMAT) return;
    int loc = gid & (NMAT - 1);
    float v = ((loc >> 7) == (loc & 127)) ? 1.f : 0.f;
    float2 e = make_float2(v, 0.f);
    Acc[gid] = e;
    Term[gid] = e;
}

// ---------------------------------------------------------------------------
// 5) Generic complex 128x128 matmul, batched over blockIdx.z via strides
// ---------------------------------------------------------------------------
__global__ void zgemm128_kernel(float2* __restrict__ C,
                                const float2* __restrict__ A,
                                const float2* __restrict__ B,
                                int sC, int sA, int sB) {
    int z = blockIdx.z;
    C += (size_t)z * sC; A += (size_t)z * sA; B += (size_t)z * sB;
    int j = blockIdx.x * 16 + threadIdx.x;
    int i = blockIdx.y * 16 + threadIdx.y;
    const float2* arow = A + (size_t)i * DIMQ;
    float re = 0.f, im = 0.f;
    for (int k = 0; k < DIMQ; ++k) {
        float2 a = arow[k];
        float2 b = B[(size_t)k * DIMQ + j];
        re = fmaf(a.x, b.x, re); re = fmaf(-a.y, b.y, re);
        im = fmaf(a.x, b.y, im); im = fmaf(a.y, b.x, im);
    }
    C[(size_t)i * DIMQ + j] = make_float2(re, im);
}

// ---------------------------------------------------------------------------
// 6) Taylor step: Tn = (To @ As) * invk ; Acc += Tn     (per-t, stride NMAT)
// ---------------------------------------------------------------------------
__global__ void ztaylor128_kernel(float2* __restrict__ Tn, float2* __restrict__ Acc,
                                  const float2* __restrict__ To,
                                  const float2* __restrict__ As, float invk) {
    size_t off = (size_t)blockIdx.z * NMAT;
    Tn += off; Acc += off; To += off; As += off;
    int j = blockIdx.x * 16 + threadIdx.x;
    int i = blockIdx.y * 16 + threadIdx.y;
    const float2* arow = To + (size_t)i * DIMQ;
    float re = 0.f, im = 0.f;
    for (int k = 0; k < DIMQ; ++k) {
        float2 a = arow[k];
        float2 b = As[(size_t)k * DIMQ + j];
        re = fmaf(a.x, b.x, re); re = fmaf(-a.y, b.y, re);
        im = fmaf(a.x, b.y, im); im = fmaf(a.y, b.x, im);
    }
    re *= invk; im *= invk;
    int idx = i * DIMQ + j;
    Tn[idx] = make_float2(re, im);
    float2 acc = Acc[idx];
    Acc[idx] = make_float2(acc.x + re, acc.y + im);
}

// ---------------------------------------------------------------------------
// 7) Pack even columns of Ufin to f16:  Ue[t][i][k] = Ufin[t][i][2k]
// ---------------------------------------------------------------------------
__global__ void pack_ue_kernel(_Float16* __restrict__ Uer, _Float16* __restrict__ Uei,
                               const float2* __restrict__ Ufin) {
    int gid = blockIdx.x * 256 + threadIdx.x;
    if (gid >= TQ * DIMQ * 64) return;
    int k = gid & 63;
    int i = (gid >> 6) & 127;
    int t = gid >> 13;
    float2 u = Ufin[(size_t)t * NMAT + i * DIMQ + 2 * k];
    Uer[gid] = (_Float16)u.x;
    Uei[gid] = (_Float16)u.y;
}

// ---------------------------------------------------------------------------
// 8) Fused WMMA kernel: st = state @ Ue^T (re+im), probs, signed sum, tanh head
//    One wave handles 16 batch rows; 8 waves/block share the 32KB LDS U tile.
// ---------------------------------------------------------------------------
__global__ void qp_fused_kernel(const float* __restrict__ state,
                                const _Float16* __restrict__ Uer,
                                const _Float16* __restrict__ Uei,
                                const float* __restrict__ Wm,
                                const float* __restrict__ bv,
                                float* __restrict__ out) {
    __shared__ __align__(32) _Float16 sh_re[DIMQ * 64];
    __shared__ __align__(32) _Float16 sh_im[DIMQ * 64];

    int tid  = threadIdx.x;
    int lane = tid & 31;
    int wave = tid >> 5;
    int g    = lane >> 4;      // half-wave group (0/1)
    int n    = lane & 15;      // N column within 16-wide tile / A row M
    int row_base = (blockIdx.x * 8 + wave) * 16;

    // A fragments (16x32 f16, K=0..31 and K=32..63), contiguous K order matching B:
    // lane m=n, lo lanes K 0..15, hi lanes K 16..31
    const float* srow = state + (size_t)(row_base + n) * 64;
    v16h aA, aB;
#pragma unroll
    for (int i = 0; i < 8; ++i) {
        aA[i]     = (_Float16)srow[g * 16 + i];
        aA[i + 8] = (_Float16)srow[g * 16 + 8 + i];
        aB[i]     = (_Float16)srow[32 + g * 16 + i];
        aB[i + 8] = (_Float16)srow[32 + g * 16 + 8 + i];
    }

    float sgn = (n & 1) ? -1.0f : 1.0f;   // exp0 - exp1 pairing over i
    float r_t[TQ][8];

    for (int t = 0; t < TQ; ++t) {
        __syncthreads();
        // stage 128x64 re + im halves (32 KB) into LDS, shared by 8 waves
        const uint4* gre = (const uint4*)(Uer + (size_t)t * DIMQ * 64);
        const uint4* gim = (const uint4*)(Uei + (size_t)t * DIMQ * 64);
        uint4* sre = (uint4*)sh_re;
        uint4* sim = (uint4*)sh_im;
        for (int i = tid; i < (DIMQ * 64) / 8; i += 256) {
            sre[i] = gre[i];
            sim[i] = gim[i];
        }
        __syncthreads();
        if (t + 1 < TQ) {   // global_prefetch_b8 of next tile
            __builtin_prefetch(Uer + (size_t)(t + 1) * DIMQ * 64, 0, 1);
            __builtin_prefetch(Uei + (size_t)(t + 1) * DIMQ * 64, 0, 1);
        }

        float racc[8];
#pragma unroll
        for (int v = 0; v < 8; ++v) racc[v] = 0.f;

#pragma unroll
        for (int c = 0; c < 8; ++c) {          // 8 N-chunks of the 128 outputs
            int col = c * 16 + n;
            // B fragments (32x16 f16): lane = column, lo lanes K 0..15, hi 16..31
            const v16h* bre0 = (const v16h*)(sh_re + col * 64 + g * 16);
            const v16h* bre1 = (const v16h*)(sh_re + col * 64 + 32 + g * 16);
            const v16h* bim0 = (const v16h*)(sh_im + col * 64 + g * 16);
            const v16h* bim1 = (const v16h*)(sh_im + col * 64 + 32 + g * 16);

            v8f cre = {};
            cre = __builtin_amdgcn_wmma_f32_16x16x32_f16(false, aA, false, *bre0,
                                                         (short)0, cre, false, false);
            cre = __builtin_amdgcn_wmma_f32_16x16x32_f16(false, aB, false, *bre1,
                                                         (short)0, cre, false, false);
            v8f cim = {};
            cim = __builtin_amdgcn_wmma_f32_16x16x32_f16(false, aA, false, *bim0,
                                                         (short)0, cim, false, false);
            cim = __builtin_amdgcn_wmma_f32_16x16x32_f16(false, aB, false, *bim1,
                                                         (short)0, cim, false, false);
#pragma unroll
            for (int v = 0; v < 8; ++v)
                racc[v] += sgn * (cre[v] * cre[v] + cim[v] * cim[v]);
        }
        // reduce across the 16 N-lanes within each half-wave group
#pragma unroll
        for (int v = 0; v < 8; ++v) {
            float s = racc[v];
            s += __shfl_xor(s, 1, 32);
            s += __shfl_xor(s, 2, 32);
            s += __shfl_xor(s, 4, 32);
            s += __shfl_xor(s, 8, 32);
            r_t[t][v] = s;
        }
    }

    // epilogue: out[b] = tanh(r @ W^T + b); lanes n==0 own 8 rows each
    if (n == 0) {
        float W0[TQ], W1[TQ];
#pragma unroll
        for (int t = 0; t < TQ; ++t) { W0[t] = Wm[t]; W1[t] = Wm[TQ + t]; }
        float b0 = bv[0], b1 = bv[1];
#pragma unroll
        for (int v = 0; v < 8; ++v) {
            int row = row_base + v + 8 * g;
            float y0 = b0, y1 = b1;
#pragma unroll
            for (int t = 0; t < TQ; ++t) {
                y0 = fmaf(r_t[t][v], W0[t], y0);
                y1 = fmaf(r_t[t][v], W1[t], y1);
            }
            out[row * 2 + 0] = tanhf(y0);
            out[row * 2 + 1] = tanhf(y1);
        }
    }
}

// ---------------------------------------------------------------------------
// Host orchestration
// ---------------------------------------------------------------------------
extern "C" void kernel_launch(void* const* d_in, const int* in_sizes, int n_in,
                              void* d_out, int out_size, void* d_ws, size_t ws_size,
                              hipStream_t stream) {
    const float* state = (const float*)d_in[0];   // (B, 64)
    const float* params = (const float*)d_in[1];  // (3, L, 2, N)
    const float* m1 = (const float*)d_in[2];
    const float* m2 = (const float*)d_in[3];
    const float* m3 = (const float*)d_in[4];
    const float* Wm = (const float*)d_in[5];      // (2, 5)
    const float* bv = (const float*)d_in[6];      // (2,)
    float* out = (float*)d_out;
    int B = in_sizes[0] / 64;                     // 32768

    // workspace carve-out (256B aligned slabs)
    char* w = (char*)d_ws;
    auto alloc = [&](size_t bytes) -> void* {
        void* p = (void*)w;
        w += (bytes + 255) & ~(size_t)255;
        return p;
    };
    float2* Gg  = (float2*)alloc((size_t)LQ * 2 * NQ * 4 * sizeof(float2));
    float2* Uu  = (float2*)alloc((size_t)8 * NMAT * sizeof(float2));   // U1/U2 per layer
    float2* As  = (float2*)alloc((size_t)TQ * NMAT * sizeof(float2));
    float2* ACC = (float2*)alloc((size_t)TQ * NMAT * sizeof(float2));
    float2* TA  = (float2*)alloc((size_t)TQ * NMAT * sizeof(float2));
    float2* TB  = (float2*)alloc((size_t)TQ * NMAT * sizeof(float2));
    float2* Xa  = (float2*)alloc((size_t)TQ * NMAT * sizeof(float2));
    float2* Xb  = (float2*)alloc((size_t)TQ * NMAT * sizeof(float2));
    _Float16* Uer = (_Float16*)alloc((size_t)TQ * DIMQ * 64 * sizeof(_Float16));
    _Float16* Uei = (_Float16*)alloc((size_t)TQ * DIMQ * 64 * sizeof(_Float16));

    dim3 tb2(16, 16);
    dim3 gT(8, 8, TQ);

    // ---- setup: As, gates, kron unitaries, expm init
    build_as_kernel<<<NMAT / 256, 256, 0, stream>>>(As, m1, m2, m3);
    build_gates_kernel<<<1, 64, 0, stream>>>(Gg, params);
    build_u_kernel<<<dim3(8, 8, 8), tb2, 0, stream>>>(Uu, Gg);
    zinit_eye_kernel<<<(TQ * NMAT + 255) / 256, 256, 0, stream>>>(ACC, TA);

    // ---- expm(-i t M): Taylor (12 terms), then square 8x
    float2* tc = TA;
    float2* tn = TB;
    for (int k = 1; k <= NTERMS; ++k) {
        ztaylor128_kernel<<<gT, tb2, 0, stream>>>(tn, ACC, tc, As, 1.0f / (float)k);
        float2* tmp = tc; tc = tn; tn = tmp;
    }
    float2* cur = ACC;
    float2* oth = TA;
    for (int s = 0; s < SCALE_POW; ++s) {   // even count -> result lands back in ACC
        zgemm128_kernel<<<gT, tb2, 0, stream>>>(oth, cur, cur, NMAT, NMAT, NMAT);
        float2* tmp = cur; cur = oth; oth = tmp;
    }
    float2* H = cur;   // == ACC

    // ---- layer chain: Ufin = U2[3] H U1[3] ... U2[0] H U1[0]   (11 zgemms)
    zgemm128_kernel<<<gT, tb2, 0, stream>>>(Xa, H, Uu + 0 * NMAT, NMAT, NMAT, 0); // H @ U1[0]
    zgemm128_kernel<<<gT, tb2, 0, stream>>>(Xb, Uu + 1 * NMAT, Xa, NMAT, 0, NMAT); // U2[0]
    zgemm128_kernel<<<gT, tb2, 0, stream>>>(Xa, Uu + 2 * NMAT, Xb, NMAT, 0, NMAT); // U1[1]
    zgemm128_kernel<<<gT, tb2, 0, stream>>>(Xb, H, Xa, NMAT, NMAT, NMAT);          // H
    zgemm128_kernel<<<gT, tb2, 0, stream>>>(Xa, Uu + 3 * NMAT, Xb, NMAT, 0, NMAT); // U2[1]
    zgemm128_kernel<<<gT, tb2, 0, stream>>>(Xb, Uu + 4 * NMAT, Xa, NMAT, 0, NMAT); // U1[2]
    zgemm128_kernel<<<gT, tb2, 0, stream>>>(Xa, H, Xb, NMAT, NMAT, NMAT);          // H
    zgemm128_kernel<<<gT, tb2, 0, stream>>>(Xb, Uu + 5 * NMAT, Xa, NMAT, 0, NMAT); // U2[2]
    zgemm128_kernel<<<gT, tb2, 0, stream>>>(Xa, Uu + 6 * NMAT, Xb, NMAT, 0, NMAT); // U1[3]
    zgemm128_kernel<<<gT, tb2, 0, stream>>>(Xb, H, Xa, NMAT, NMAT, NMAT);          // H
    zgemm128_kernel<<<gT, tb2, 0, stream>>>(Xa, Uu + 7 * NMAT, Xb, NMAT, 0, NMAT); // U2[3] -> Ufin

    // ---- pack even columns to f16 and run the fused WMMA pass
    pack_ue_kernel<<<(TQ * DIMQ * 64 + 255) / 256, 256, 0, stream>>>(Uer, Uei, Xa);
    qp_fused_kernel<<<B / 128, 256, 0, stream>>>(state, Uer, Uei, Wm, bv, out);

    (void)n_in; (void)out_size; (void)ws_size;
}
